// AttentionCell_11793980194907
// MI455X (gfx1250) — matrix operand order repaired and checked by reading, not compiled, not run on hardware
//
#include <hip/hip_runtime.h>
#include <hip/hip_bf16.h>
#include <math.h>

typedef __attribute__((ext_vector_type(16))) __bf16 bf16x16;
typedef __attribute__((ext_vector_type(8)))  __bf16 bf16x8;
typedef __attribute__((ext_vector_type(4)))  __bf16 bf16x4;
typedef __attribute__((ext_vector_type(8)))  float  f32x8;
typedef __attribute__((ext_vector_type(4)))  int    v4i;

#define T_DIM 256
#define B_DIM 256
#define C_DIM 512
#define H_DIM 256
#define E_DIM 128
#define X_DIM (C_DIM + E_DIM)   /* 640 */
#define G_DIM (3 * H_DIM)       /* 768 */

#define GLOBAL_AS __attribute__((address_space(1)))
#define LDS_AS    __attribute__((address_space(3)))

#if defined(__HIP_DEVICE_COMPILE__) && __has_builtin(__builtin_amdgcn_global_load_async_to_lds_b128)
#define HAVE_ASYNC_LDS 1
#else
#define HAVE_ASYNC_LDS 0
#endif

// ---------------------------------------------------------------------------
// WMMA helpers: D = A(16x32 bf16) * B(32x16 bf16) + C(16x16 f32), wave32.
// ---------------------------------------------------------------------------
__device__ __forceinline__ f32x8 wmma_bf16f32(bf16x16 a, bf16x16 b, f32x8 c) {
  return __builtin_amdgcn_wmma_f32_16x16x32_bf16(
      /*neg_a=*/false, a, /*neg_b=*/false, b,
      /*c_mod=*/(short)0, c, /*reuse_a=*/false, /*reuse_b=*/false);
}

// A fragment (16x32, 16-bit), contiguous k order used here (same as B):
//   lanes 0-15 : row M=lane,    K = [k0..k0+15]
//   lanes 16-31: row M=lane-16, K = [k0+16..k0+31]
__device__ __forceinline__ bf16x16 load_a_frag(const __bf16* rowbase, int stride,
                                               int k0, int lane) {
  const int half = lane >> 4;
  const int r    = lane & 15;
  const __bf16* p = rowbase + (size_t)r * stride + k0 + half * 16;
  bf16x8 lo = *(const bf16x8*)p;         // K = k0+16h .. +7
  bf16x8 hi = *(const bf16x8*)(p + 8);   // K = k0+16h+8 .. +7
  bf16x16 a;
#pragma unroll
  for (int i = 0; i < 8; ++i) { a[i] = lo[i]; a[8 + i] = hi[i]; }
  return a;
}

// B fragment (32x16, 16-bit): lane holds column N=n0+(lane&15);
//   lanes 0-15 get K=[k0..k0+15], lanes 16-31 get K=[k0+16..k0+31].
__device__ __forceinline__ bf16x16 load_b_frag(const __bf16* Bt, int stride,
                                               int n0, int k0, int lane) {
  const int half = lane >> 4;
  const int cc   = lane & 15;
  const __bf16* p = Bt + (size_t)(n0 + cc) * stride + k0 + half * 16;
  return *(const bf16x16*)p;
}

// ---------------------------------------------------------------------------
// f32 -> bf16 conversion (grid-stride)
// ---------------------------------------------------------------------------
__global__ void f2bf_kernel(const float* __restrict__ in, __bf16* __restrict__ out, int n) {
  for (int i = blockIdx.x * blockDim.x + threadIdx.x; i < n; i += gridDim.x * blockDim.x)
    out[i] = (__bf16)in[i];
}

// cur_embeddings -> bf16 columns [512..639] of x
__global__ void emb_to_x_kernel(const float* __restrict__ emb, __bf16* __restrict__ xbf) {
  int i = blockIdx.x * blockDim.x + threadIdx.x;
  if (i < B_DIM * E_DIM) {
    int b = i >> 7, e = i & (E_DIM - 1);
    xbf[(size_t)b * X_DIM + C_DIM + e] = (__bf16)emb[i];
  }
}

// ---------------------------------------------------------------------------
// Generic C[M,N] = A[M,K](bf16) * Bt[N,K]^T(bf16) + bias[N], one wave per 16x16 tile
// ---------------------------------------------------------------------------
__global__ void gemm_bf16_kernel(const __bf16* __restrict__ A, const __bf16* __restrict__ Bt,
                                 const float* __restrict__ bias, float* __restrict__ Cm,
                                 int M, int N, int K) {
  const int lane = threadIdx.x & 31;
  const int tile = blockIdx.x * (blockDim.x >> 5) + (threadIdx.x >> 5);
  const int ntn  = N >> 4;
  if (tile >= (M >> 4) * ntn) return;  // wave-uniform
  const int m0 = (tile / ntn) << 4;
  const int n0 = (tile % ntn) << 4;
  const __bf16* arow = A + (size_t)m0 * K;
  f32x8 acc = {};
  for (int k0 = 0; k0 < K; k0 += 32) {
    bf16x16 a = load_a_frag(arow, K, k0, lane);
    bf16x16 b = load_b_frag(Bt, K, n0, k0, lane);
    acc = wmma_bf16f32(a, b, acc);
  }
  const int half = lane >> 4, cc = lane & 15;
  const float bv = bias ? bias[n0 + cc] : 0.0f;
#pragma unroll
  for (int j = 0; j < 8; ++j)
    Cm[(size_t)(m0 + j + 8 * half) * N + n0 + cc] = acc[j] + bv;
}

// ---------------------------------------------------------------------------
// Fused attention scores:
//   emition[t,b] = sum_h tanh( (feats @ W_i2h^T)[t,b,h] + hidden_proj[b,h] ) * w_score[h]
// One wave per 16-row M-tile of flattened (t,b). feats staged into LDS (f32->bf16),
// via async-DMA (ASYNCcnt path) when available, else through VGPRs.
// 2 waves / 64-thread block; LDS: 2*16KB bf16 tiles (+ 2*16KB f32 scratch if async).
// ---------------------------------------------------------------------------
__global__ void attn_scores_kernel(const float* __restrict__ feats,
                                   const __bf16* __restrict__ Wi2h,
                                   const float* __restrict__ hidden_proj,
                                   const float* __restrict__ w_score,
                                   float* __restrict__ emition) {
  __shared__ __bf16 tileb[2][16 * C_DIM];       // 2 x 16KB
#if HAVE_ASYNC_LDS
  __shared__ float  stagef[2][8 * C_DIM];       // 2 x 16KB f32 half-tile scratch
#endif
  const int wv = threadIdx.x >> 5, lane = threadIdx.x & 31;
  const int mt = blockIdx.x * 2 + wv;
  const int m0 = mt << 4;  // flat row index into feats[T*B, C]
  const float* src = feats + (size_t)m0 * C_DIM;
  __bf16* dst = tileb[wv];

#if HAVE_ASYNC_LDS
  float* stg = stagef[wv];
  // Two phases of 8 rows: async DMA global->LDS (f32), then convert to bf16 tile.
  for (int ph = 0; ph < 2; ++ph) {
    const float* s = src + ph * (8 * C_DIM);
    // wait until previous phase's LDS reads of stg are done before DMA overwrites it
    asm volatile("s_wait_dscnt 0x0" ::: "memory");
#pragma unroll
    for (int it = 0; it < 32; ++it) {  // 32 x (32 lanes x 16B) = 16KB
      __builtin_amdgcn_global_load_async_to_lds_b128(
          (GLOBAL_AS v4i*)(s + it * 128 + lane * 4),
          (LDS_AS v4i*)(stg + it * 128 + lane * 4), 0, 0);
    }
    asm volatile("s_wait_asynccnt 0x0" ::: "memory");
    __bf16* d = dst + ph * (8 * C_DIM);
    for (int i = lane; i < (8 * C_DIM) / 4; i += 32) {
      float4 f = ((const float4*)stg)[i];
      bf16x4 o;
      o[0] = (__bf16)f.x; o[1] = (__bf16)f.y; o[2] = (__bf16)f.z; o[3] = (__bf16)f.w;
      *(bf16x4*)(d + 4 * i) = o;
    }
  }
#else
  for (int i = lane; i < (16 * C_DIM) / 4; i += 32) {
    float4 f = ((const float4*)src)[i];
    bf16x4 o;
    o[0] = (__bf16)f.x; o[1] = (__bf16)f.y; o[2] = (__bf16)f.z; o[3] = (__bf16)f.w;
    *(bf16x4*)(dst + 4 * i) = o;
  }
#endif
  __syncthreads();

  const int half = lane >> 4, cc = lane & 15;
  const int b0 = m0 & (B_DIM - 1);  // b of first row (tile has constant t)
  float esum[8];
#pragma unroll
  for (int j = 0; j < 8; ++j) esum[j] = 0.0f;

  for (int nt = 0; nt < H_DIM / 16; ++nt) {
    const int n0 = nt << 4;
    f32x8 acc = {};
#pragma unroll 4
    for (int k0 = 0; k0 < C_DIM; k0 += 32) {
      bf16x16 a = load_a_frag(dst, C_DIM, k0, lane);
      bf16x16 b = load_b_frag(Wi2h, C_DIM, n0, k0, lane);
      acc = wmma_bf16f32(a, b, acc);
    }
    const int h = n0 + cc;
    const float wsv = w_score[h];
#pragma unroll
    for (int j = 0; j < 8; ++j) {
      const int bb = b0 + j + 8 * half;  // row's batch index
      float v = acc[j] + hidden_proj[(size_t)bb * H_DIM + h];
      esum[j] += tanhf(v) * wsv;
    }
  }
  // Reduce each row's partial over its 16 lanes (halves are independent rows).
#pragma unroll
  for (int j = 0; j < 8; ++j) {
    float v = esum[j];
#pragma unroll
    for (int msk = 1; msk < 16; msk <<= 1) v += __shfl_xor(v, msk, 16);
    if (cc == 0) emition[m0 + j + 8 * half] = v;
  }
}

// ---------------------------------------------------------------------------
// Softmax over T (axis 0) per batch column; one wave per b.
// ---------------------------------------------------------------------------
__global__ void softmax_t_kernel(const float* __restrict__ em, float* __restrict__ alpha) {
  const int lane = threadIdx.x & 31;
  const int b = blockIdx.x * (blockDim.x >> 5) + (threadIdx.x >> 5);
  float v[T_DIM / 32];
  float mx = -1e30f;
#pragma unroll
  for (int i = 0; i < T_DIM / 32; ++i) {
    v[i] = em[(size_t)(lane + 32 * i) * B_DIM + b];
    mx = fmaxf(mx, v[i]);
  }
#pragma unroll
  for (int msk = 1; msk < 32; msk <<= 1) mx = fmaxf(mx, __shfl_xor(mx, msk, 32));
  float s = 0.0f;
#pragma unroll
  for (int i = 0; i < T_DIM / 32; ++i) { v[i] = __expf(v[i] - mx); s += v[i]; }
#pragma unroll
  for (int msk = 1; msk < 32; msk <<= 1) s += __shfl_xor(s, msk, 32);
  const float inv = 1.0f / s;
#pragma unroll
  for (int i = 0; i < T_DIM / 32; ++i)
    alpha[(size_t)(lane + 32 * i) * B_DIM + b] = v[i] * inv;
}

// ---------------------------------------------------------------------------
// context[b,c] = sum_t feats[t,b,c] * alpha[t,b]; written as bf16 into x[:, :512].
// feats should still be L2-resident from the scores pass (134MB < 192MB L2).
// ---------------------------------------------------------------------------
__global__ void context_kernel(const float* __restrict__ feats, const float* __restrict__ alpha,
                               __bf16* __restrict__ xbf) {
  const int idx = blockIdx.x * blockDim.x + threadIdx.x;  // B*C threads
  const int b = idx >> 9, c = idx & (C_DIM - 1);
  float acc = 0.0f;
  for (int t = 0; t < T_DIM; ++t) {
    const int tp = (t + 8 < T_DIM) ? t + 8 : t;  // prefetch next slice (L2 warm)
    __builtin_prefetch(&feats[((size_t)tp * B_DIM + b) * C_DIM + c], 0, 0);
    acc = fmaf(feats[((size_t)t * B_DIM + b) * C_DIM + c], alpha[t * B_DIM + b], acc);
  }
  xbf[(size_t)b * X_DIM + c] = (__bf16)acc;
}

// ---------------------------------------------------------------------------
// GRU gate elementwise: gates ordered (r, z, n), PyTorch GRUCell semantics.
// ---------------------------------------------------------------------------
__device__ __forceinline__ float sigmoidf_(float x) { return 1.0f / (1.0f + __expf(-x)); }

__global__ void gru_kernel(const float* __restrict__ gi, const float* __restrict__ gh,
                           const float* __restrict__ prev, float* __restrict__ out) {
  const int idx = blockIdx.x * blockDim.x + threadIdx.x;  // B*H
  const int b = idx >> 8, h = idx & (H_DIM - 1);
  const float* gib = gi + (size_t)b * G_DIM;
  const float* ghb = gh + (size_t)b * G_DIM;
  const float r = sigmoidf_(gib[h] + ghb[h]);
  const float z = sigmoidf_(gib[H_DIM + h] + ghb[H_DIM + h]);
  const float n = tanhf(gib[2 * H_DIM + h] + r * ghb[2 * H_DIM + h]);
  out[idx] = (1.0f - z) * n + z * prev[idx];
}

// ---------------------------------------------------------------------------
extern "C" void kernel_launch(void* const* d_in, const int* in_sizes, int n_in,
                              void* d_out, int out_size, void* d_ws, size_t ws_size,
                              hipStream_t stream) {
  (void)in_sizes; (void)n_in; (void)out_size; (void)ws_size;
  const float* prev_hidden = (const float*)d_in[0];
  const float* feats       = (const float*)d_in[1];
  const float* cur_emb     = (const float*)d_in[2];
  const float* W_i2h       = (const float*)d_in[3];
  const float* W_h2h       = (const float*)d_in[4];
  const float* b_h2h       = (const float*)d_in[5];
  const float* w_score     = (const float*)d_in[6];
  const float* W_ih        = (const float*)d_in[7];
  const float* W_hh        = (const float*)d_in[8];
  const float* b_ih        = (const float*)d_in[9];
  const float* b_hh        = (const float*)d_in[10];

  float* out_hidden = (float*)d_out;                    // [B,H]
  float* out_alpha  = (float*)d_out + B_DIM * H_DIM;    // [T,B]

  char* ws = (char*)d_ws;
  size_t off = 0;
  auto alloc = [&](size_t bytes) -> void* {
    void* p = ws + off;
    off = (off + bytes + 255) & ~(size_t)255;
    return p;
  };
  __bf16* wi2h_bf = (__bf16*)alloc((size_t)H_DIM * C_DIM * 2);
  __bf16* wh2h_bf = (__bf16*)alloc((size_t)H_DIM * H_DIM * 2);
  __bf16* wih_bf  = (__bf16*)alloc((size_t)G_DIM * X_DIM * 2);
  __bf16* whh_bf  = (__bf16*)alloc((size_t)G_DIM * H_DIM * 2);
  __bf16* prev_bf = (__bf16*)alloc((size_t)B_DIM * H_DIM * 2);
  __bf16* x_bf    = (__bf16*)alloc((size_t)B_DIM * X_DIM * 2);
  float* hidden_proj = (float*)alloc((size_t)B_DIM * H_DIM * 4);
  float* emition     = (float*)alloc((size_t)T_DIM * B_DIM * 4);
  float* gi          = (float*)alloc((size_t)B_DIM * G_DIM * 4);
  float* gh          = (float*)alloc((size_t)B_DIM * G_DIM * 4);

  // 1) weight / state conversions to bf16
  f2bf_kernel<<<256, 256, 0, stream>>>(W_i2h, wi2h_bf, H_DIM * C_DIM);
  f2bf_kernel<<<128, 256, 0, stream>>>(W_h2h, wh2h_bf, H_DIM * H_DIM);
  f2bf_kernel<<<512, 256, 0, stream>>>(W_ih, wih_bf, G_DIM * X_DIM);
  f2bf_kernel<<<256, 256, 0, stream>>>(W_hh, whh_bf, G_DIM * H_DIM);
  f2bf_kernel<<<128, 256, 0, stream>>>(prev_hidden, prev_bf, B_DIM * H_DIM);
  emb_to_x_kernel<<<(B_DIM * E_DIM) / 256, 256, 0, stream>>>(cur_emb, x_bf);

  // 2) hidden_proj = prev_hidden @ W_h2h^T + b_h2h   [256,256]
  gemm_bf16_kernel<<<(16 * 16) / 4, 128, 0, stream>>>(prev_bf, wh2h_bf, b_h2h,
                                                      hidden_proj, B_DIM, H_DIM, H_DIM);

  // 3) fused big GEMM + tanh + score-dot -> emition [T,B]   (2 waves / 64-thread block)
  attn_scores_kernel<<<(T_DIM * B_DIM / 16) / 2, 64, 0, stream>>>(
      feats, wi2h_bf, hidden_proj, w_score, emition);

  // 4) softmax over T -> alpha (second output)
  softmax_t_kernel<<<B_DIM / 4, 128, 0, stream>>>(emition, out_alpha);

  // 5) context -> x[:, :512] (bf16)
  context_kernel<<<(B_DIM * C_DIM) / 256, 256, 0, stream>>>(feats, out_alpha, x_bf);

  // 6) GRU GEMMs: gi = x @ W_ih^T + b_ih ; gh = prev @ W_hh^T + b_hh
  gemm_bf16_kernel<<<(16 * 48) / 4, 128, 0, stream>>>(x_bf, wih_bf, b_ih,
                                                      gi, B_DIM, G_DIM, X_DIM);
  gemm_bf16_kernel<<<(16 * 48) / 4, 128, 0, stream>>>(prev_bf, whh_bf, b_hh,
                                                      gh, B_DIM, G_DIM, H_DIM);

  // 7) GRU gate math -> cur_hidden (first output)
  gru_kernel<<<(B_DIM * H_DIM) / 256, 256, 0, stream>>>(gi, gh, prev_hidden, out_hidden);
}
